// QuantPINN_NLSE_54305566490841
// MI455X (gfx1250) — 8-bit matrix operand order repaired and checked by reading, compiled, not run on hardware
//
#include <hip/hip_runtime.h>


#define QMAXF 127.0f
#define ACT_SCALE (1.0f / 127.0f)
#define EPSF 1e-8f

typedef __attribute__((ext_vector_type(8))) int v8i;

__device__ __forceinline__ unsigned pack4(int a, int b, int c, int d) {
  return (unsigned)(a & 255) | ((unsigned)(b & 255) << 8) |
         ((unsigned)(c & 255) << 16) | ((unsigned)(d & 255) << 24);
}

__device__ __forceinline__ int qcode(float x, float scale) {
  float r = rintf(x / scale);
  r = fminf(fmaxf(r, -127.0f), 127.0f);
  return (int)r;
}

__global__ __launch_bounds__(256) void qpinn_maxabs(const float* __restrict__ z,
                                                    const float* __restrict__ t,
                                                    int N,
                                                    float* __restrict__ partial) {
  __shared__ float red[256];
  float m = 0.0f;
  for (int i = blockIdx.x * blockDim.x + threadIdx.x; i < N;
       i += gridDim.x * blockDim.x) {
    m = fmaxf(m, fmaxf(fabsf(z[i]), fabsf(t[i])));
  }
  red[threadIdx.x] = m;
  __syncthreads();
  for (int o = 128; o > 0; o >>= 1) {
    if ((int)threadIdx.x < o)
      red[threadIdx.x] = fmaxf(red[threadIdx.x], red[threadIdx.x + o]);
    __syncthreads();
  }
  if (threadIdx.x == 0) partial[blockIdx.x] = red[0];
}

__global__ __launch_bounds__(256) void qpinn_prep(
    const float* __restrict__ W1, const float* __restrict__ b1,
    const float* __restrict__ W2, const float* __restrict__ b2,
    const float* __restrict__ W3, const float* __restrict__ b3,
    const float* __restrict__ W4, const float* __restrict__ b4,
    const float* __restrict__ Wo, const float* __restrict__ bo,
    const float* __restrict__ partial, int nPartial,
    float* __restrict__ scales, float* __restrict__ bq,
    unsigned* __restrict__ packed) {
  __shared__ float red[256];
  __shared__ float sh_in_scale;
  const int tid = threadIdx.x;

  float m = 0.0f;
  for (int i = tid; i < nPartial; i += 256) m = fmaxf(m, partial[i]);
  red[tid] = m;
  __syncthreads();
  for (int o = 128; o > 0; o >>= 1) {
    if (tid < o) red[tid] = fmaxf(red[tid], red[tid + o]);
    __syncthreads();
  }
  if (tid == 0) {
    sh_in_scale = fmaxf(red[0], EPSF) / QMAXF;
    scales[5] = sh_in_scale;
  }
  __syncthreads();

  const float* Ws[5] = {W1, W2, W3, W4, Wo};
  const float* bs[5] = {b1, b2, b3, b4, bo};
  const int inD[5] = {2, 50, 50, 50, 50};
  const int outD[5] = {50, 50, 50, 50, 2};
  const int ntl[5] = {4, 4, 4, 4, 1};
  const int poff[5] = {0, 1024, 2048, 3072, 4096};

  for (int l = 0; l < 5; ++l) {
    const int wn = inD[l] * outD[l];
    float wm = 0.0f;
    for (int i = tid; i < wn; i += 256) wm = fmaxf(wm, fabsf(Ws[l][i]));
    red[tid] = wm;
    __syncthreads();
    for (int o = 128; o > 0; o >>= 1) {
      if (tid < o) red[tid] = fmaxf(red[tid], red[tid + o]);
      __syncthreads();
    }
    const float wsc = fmaxf(red[0], EPSF) / QMAXF;
    const float prev = (l == 0) ? sh_in_scale : ACT_SCALE;
    const float sl = prev * wsc;
    if (tid == 0) scales[l] = sl;

    if (tid < 64) {
      float q = 0.0f;
      if (tid < outD[l]) {
        float r = rintf(bs[l][tid] / sl);
        r = fminf(fmaxf(r, -127.0f), 127.0f);
        q = r * sl;
      }
      bq[l * 64 + tid] = q;
    }

    const int tot = ntl[l] * 256;
    for (int idx = tid; idx < tot; idx += 256) {
      const int t = idx >> 8;
      const int rem = idx & 255;
      const int lane = rem >> 3;
      const int v = rem & 7;
      const int h = lane >> 4;
      const int mrow = t * 16 + (lane & 15);
      unsigned w = 0;
#pragma unroll
      for (int byt = 0; byt < 4; ++byt) {
        const int k = 32 * (v >> 2) + 16 * ((v >> 1) & 1) + 8 * h + 4 * (v & 1) + byt;
        int code = 0;
        if (mrow < outD[l] && k < inD[l]) {
          float r = rintf(Ws[l][mrow * inD[l] + k] / wsc);
          r = fminf(fmaxf(r, -127.0f), 127.0f);
          code = (int)r;
        }
        w |= ((unsigned)(code & 255)) << (8 * byt);
      }
      packed[poff[l] + idx] = w;
    }
    __syncthreads();
  }
}

__device__ __forceinline__ v8i qpinn_hidden(const unsigned* __restrict__ pA,
                                            float sc,
                                            const float* __restrict__ bql,
                                            v8i B, int lane) {
  const int h = lane >> 4;
  unsigned lo[4], hi[4];
#pragma unroll
  for (int t = 0; t < 4; ++t) {
    const uint4* p = (const uint4*)(pA + (t * 32 + lane) * 8);
    const uint4 a0 = p[0];
    const uint4 a1 = p[1];
    v8i A;
    A[0] = (int)a0.x; A[1] = (int)a0.y; A[2] = (int)a0.z; A[3] = (int)a0.w;
    A[4] = (int)a1.x; A[5] = (int)a1.y; A[6] = (int)a1.z; A[7] = (int)a1.w;
    v8i c = {};
    v8i d = __builtin_amdgcn_wmma_i32_16x16x64_iu8(true, A, true, B, c, false, false);
    const float4 bb0 = *(const float4*)(bql + t * 16 + h * 8);
    const float4 bb1 = *(const float4*)(bql + t * 16 + h * 8 + 4);
    const float bb[8] = {bb0.x, bb0.y, bb0.z, bb0.w, bb1.x, bb1.y, bb1.z, bb1.w};
    int cd[8];
#pragma unroll
    for (int v = 0; v < 8; ++v) {
      float hf = (float)d[v] * sc + bb[v];
      hf = fminf(fmaxf(hf, -1.0f), 1.0f);
      cd[v] = (int)rintf(hf * 127.0f);
    }
    lo[t] = pack4(cd[0], cd[1], cd[2], cd[3]);
    hi[t] = pack4(cd[4], cd[5], cd[6], cd[7]);
  }
  const unsigned rA = (unsigned)__builtin_amdgcn_ds_swizzle((int)(h ? lo[0] : lo[1]), 0x401F);
  const unsigned rB = (unsigned)__builtin_amdgcn_ds_swizzle((int)(h ? hi[0] : hi[1]), 0x401F);
  const unsigned rC = (unsigned)__builtin_amdgcn_ds_swizzle((int)(h ? lo[2] : lo[3]), 0x401F);
  const unsigned rD = (unsigned)__builtin_amdgcn_ds_swizzle((int)(h ? hi[2] : hi[3]), 0x401F);
  v8i nb;
  nb[0] = (int)lo[0];
  nb[1] = (int)hi[0];
  nb[2] = (int)lo[1];
  nb[3] = (int)hi[1];
  nb[4] = (int)lo[2];
  nb[5] = (int)hi[2];
  nb[6] = (int)lo[3];
  nb[7] = (int)hi[3];
  return nb;
}

__global__ __launch_bounds__(256) void qpinn_main(
    const float* __restrict__ z, const float* __restrict__ t,
    const float* __restrict__ scales, const float* __restrict__ bq,
    const unsigned* __restrict__ packed, float* __restrict__ out, int N) {
  const int lane = threadIdx.x & 31;
  const int wave = threadIdx.x >> 5;
  const int tile = blockIdx.x * 8 + wave;
  const int nTiles = (N + 15) >> 4;
  if (tile >= nTiles) return;

  const int m0 = tile * 16;
  const int s = lane & 15;
  const int h = lane >> 4;

  const float in_scale = scales[5];
  int m = m0 + s;
  if (m >= N) m = N - 1;
  const int qz = qcode(z[m], in_scale);
  const int qt = qcode(t[m], in_scale);
  v8i B = {};
  B[0] = (int)((h == 0) ? pack4(qz, qt, 0, 0) : 0u);

#pragma unroll
  for (int l = 0; l < 4; ++l) {
    B = qpinn_hidden(packed + 1024 * l, scales[l], bq + 64 * l, B, lane);
  }

  {
    const uint4* p = (const uint4*)(packed + 4096 + lane * 8);
    const uint4 a0 = p[0];
    const uint4 a1 = p[1];
    v8i A;
    A[0] = (int)a0.x; A[1] = (int)a0.y; A[2] = (int)a0.z; A[3] = (int)a0.w;
    A[4] = (int)a1.x; A[5] = (int)a1.y; A[6] = (int)a1.z; A[7] = (int)a1.w;
    v8i c = {};
    v8i d = __builtin_amdgcn_wmma_i32_16x16x64_iu8(true, A, true, B, c, false, false);
    const float so = scales[4];
    const float u = (float)d[0] * so + bq[4 * 64 + 0];
    const float v = (float)d[1] * so + bq[4 * 64 + 1];
    const int mr = m0 + s;
    if (h == 0 && mr < N) {
      float2 o;
      o.x = u;
      o.y = v;
      reinterpret_cast<float2*>(out)[mr] = o;
    }
  }
}

extern "C" void kernel_launch(void* const* d_in, const int* in_sizes, int n_in,
                              void* d_out, int out_size, void* d_ws, size_t ws_size,
                              hipStream_t stream) {
  const float* z = (const float*)d_in[0];
  const float* t = (const float*)d_in[1];
  const float* W1 = (const float*)d_in[2];
  const float* b1 = (const float*)d_in[3];
  const float* W2 = (const float*)d_in[4];
  const float* b2 = (const float*)d_in[5];
  const float* W3 = (const float*)d_in[6];
  const float* b3 = (const float*)d_in[7];
  const float* W4 = (const float*)d_in[8];
  const float* b4 = (const float*)d_in[9];
  const float* Wo = (const float*)d_in[10];
  const float* bo = (const float*)d_in[11];
  const int N = in_sizes[0];

  float* partial = (float*)d_ws;
  float* scales = partial + 256;
  float* bq = partial + 264;
  unsigned* packed = (unsigned*)(partial + 584);

  qpinn_maxabs<<<256, 256, 0, stream>>>(z, t, N, partial);
  qpinn_prep<<<1, 256, 0, stream>>>(W1, b1, W2, b2, W3, b3, W4, b4, Wo, bo,
                                    partial, 256, scales, bq, packed);
  const int nTiles = (N + 15) / 16;
  qpinn_main<<<(nTiles + 7) / 8, 256, 0, stream>>>(z, t, scales, bq, packed,
                                                   (float*)d_out, N);
}
